// EncoderFusionLayer_69836168233417
// MI455X (gfx1250) — matrix operand order repaired and checked by reading, not compiled, not run on hardware
//
#include <hip/hip_runtime.h>
#include <math.h>

// ---------------------------------------------------------------------------
// EncoderFusionLayer for MI455X (gfx1250, wave32, WMMA bf16)
// D=512, H=16, DK=32, FF=2048, seq=512, fcp=64 -> n1=576, kv=640, B=32
// ---------------------------------------------------------------------------

#define DMODEL 512
#define NSEQ   576
#define NKV    640
#define NFF    2048
#define NBATCH 32

typedef unsigned short u16;
typedef __attribute__((ext_vector_type(16))) __bf16 v16bf;
typedef __attribute__((ext_vector_type(8)))  float  v8f;
typedef __attribute__((ext_vector_type(4)))  int    v4i_;

struct __attribute__((aligned(16))) U4 { unsigned int a, b, c, d; };

union FragBF { v16bf v; U4 q[2]; u16 h[16]; };
union FragF  { v8f  v; float f[8]; };

__device__ __forceinline__ u16 f2bf(float x) {
  union { float f; unsigned u; } c; c.f = x;
  unsigned r = c.u + 0x7FFFu + ((c.u >> 16) & 1u);   // round-to-nearest-even
  return (u16)(r >> 16);
}

// ---- CDNA5 async global->LDS path (guarded; sync fallback otherwise) ------
#if defined(__has_builtin)
#if __has_builtin(__builtin_amdgcn_global_load_async_to_lds_b128) && \
    __has_builtin(__builtin_amdgcn_s_wait_asynccnt)
#define USE_ASYNC_LDS 1
#endif
#endif

#ifdef USE_ASYNC_LDS
// builtin signature (from hipcc diagnostic): (v4i AS1*, v4i AS3*, imm, imm)
typedef __attribute__((address_space(1))) v4i_ gv4i;
typedef __attribute__((address_space(3))) v4i_ lv4i;
__device__ __forceinline__ void async_cp16(const u16* g, u16* l) {
  // 16 bytes per lane, memory -> LDS, tracked by ASYNCcnt
  __builtin_amdgcn_global_load_async_to_lds_b128((gv4i*)g, (lv4i*)l, 0, 0);
}
__device__ __forceinline__ void async_wait0() {
  __builtin_amdgcn_s_wait_asynccnt(0);
}
#endif

// ---------------------------------------------------------------------------
// Build input1 (f32+bf16) and kv_in (bf16):
//   input1 = concat(fcp[64], input[512])           -> 576 rows
//   kv_in  = concat(input1[576], extra[64])        -> 640 rows
// ---------------------------------------------------------------------------
__global__ __launch_bounds__(256) void k_build_inputs(
    const float* __restrict__ fcp, const float* __restrict__ inp,
    const float* __restrict__ extra,
    float* __restrict__ in1_f, u16* __restrict__ in1_b, u16* __restrict__ kv_b)
{
  long i = (long)blockIdx.x * 256 + threadIdx.x;
  const long total = (long)NBATCH * NKV * DMODEL;
  if (i >= total) return;
  int d = (int)(i % DMODEL);
  long t = i / DMODEL;
  int n = (int)(t % NKV);
  int b = (int)(t / NKV);
  float v;
  if (n < 64)        v = fcp  [((long)b * 64  + n)        * DMODEL + d];
  else if (n < 576)  v = inp  [((long)b * 512 + (n - 64)) * DMODEL + d];
  else               v = extra[((long)b * 64  + (n - 576))* DMODEL + d];
  u16 bv = f2bf(v);
  kv_b[i] = bv;
  if (n < 576) {
    long j = ((long)b * NSEQ + n) * DMODEL + d;
    in1_f[j] = v;
    in1_b[j] = bv;
  }
}

// pref_node = inp[:, -1, :]  (32 x 512) as bf16
__global__ __launch_bounds__(256) void k_build_pref(
    const float* __restrict__ inp, u16* __restrict__ pref_b)
{
  int i = blockIdx.x * 256 + threadIdx.x;
  if (i >= NBATCH * DMODEL) return;
  int d = i & (DMODEL - 1);
  int b = i / DMODEL;
  pref_b[i] = f2bf(inp[((long)b * 512 + 511) * DMODEL + d]);
}

// Transpose + cast a weight (K x N f32, row-major) -> Wt (N x K bf16)
__global__ __launch_bounds__(256) void k_wt(
    const float* __restrict__ W, u16* __restrict__ Wt, int K, int N)
{
  long i = (long)blockIdx.x * 256 + threadIdx.x;
  if (i >= (long)K * N) return;
  int n = (int)(i % N);
  int k = (int)(i / N);
  Wt[(long)n * K + k] = f2bf(W[i]);
}

// ---------------------------------------------------------------------------
// bf16 WMMA GEMM:  C[M,N] = A[M,K] @ Bt[N,K]^T  (+bias)(gelu)(+resid)
// 128 threads = 4 waves; block tile 64x128; wave tile 32x64 = 2x4 WMMA tiles.
// Staging: async global->LDS (ASYNCcnt) when available, else load+ds_store.
// ---------------------------------------------------------------------------
__global__ __launch_bounds__(128) void k_gemm(
    const u16* __restrict__ A, const u16* __restrict__ Bt,
    const float* __restrict__ bias, const float* __restrict__ resid,
    float* __restrict__ outF, u16* __restrict__ outB,
    int M, int N, int K, int doGelu)
{
  __shared__ __attribute__((aligned(16))) u16 As[64  * 32];   // 4 KB
  __shared__ __attribute__((aligned(16))) u16 Bs[128 * 32];   // 8 KB

  const int tid  = threadIdx.x;
  const int lane = tid & 31;
  const int wid  = tid >> 5;
  const int mBase = (wid >> 1) * 32;   // wave M offset (0/32)
  const int nBase = (wid & 1) * 64;    // wave N offset (0/64)
  const int tileM = blockIdx.y * 64;
  const int tileN = blockIdx.x * 128;

  FragF acc[2][4];
#pragma unroll
  for (int i = 0; i < 2; i++)
#pragma unroll
    for (int j = 0; j < 4; j++)
#pragma unroll
      for (int e = 0; e < 8; e++) acc[i][j].f[e] = 0.f;

  // Partial-M block (only the tiny pv GEMM): pre-zero the A tile once;
  // masked-off async lanes then leave zeros in place.
  if (tileM + 64 > M) {
#pragma unroll
    for (int p = 0; p < 2; p++) {
      U4 z; z.a = z.b = z.c = z.d = 0u;
      *(U4*)(As + (p * 128 + tid) * 8) = z;
    }
    __syncthreads();
  }

  for (int k0 = 0; k0 < K; k0 += 32) {
#ifdef USE_ASYNC_LDS
    // A tile: 64x32 bf16, one b128 per lane-slot, EXEC-masked on M bound
#pragma unroll
    for (int p = 0; p < 2; p++) {
      int t = p * 128 + tid;
      int r = t >> 2;
      int c = (t & 3) * 8;
      int gr = tileM + r;
      if (gr < M) async_cp16(A + (long)gr * K + k0 + c, As + r * 32 + c);
    }
    // B tile: 128x32 bf16 (N always a multiple of 128 here)
#pragma unroll
    for (int p = 0; p < 4; p++) {
      int t = p * 128 + tid;
      int r = t >> 2;
      int c = (t & 3) * 8;
      async_cp16(Bt + (long)(tileN + r) * K + k0 + c, Bs + r * 32 + c);
    }
    async_wait0();
    __syncthreads();
#else
#pragma unroll
    for (int p = 0; p < 2; p++) {
      int t = p * 128 + tid;
      int r = t >> 2;
      int c = (t & 3) * 8;
      int gr = tileM + r;
      U4 va; va.a = va.b = va.c = va.d = 0u;
      if (gr < M) va = *(const U4*)(A + (long)gr * K + k0 + c);
      *(U4*)(As + r * 32 + c) = va;
    }
#pragma unroll
    for (int p = 0; p < 4; p++) {
      int t = p * 128 + tid;
      int r = t >> 2;
      int c = (t & 3) * 8;
      U4 vb = *(const U4*)(Bt + (long)(tileN + r) * K + k0 + c);
      *(U4*)(Bs + r * 32 + c) = vb;
    }
    __syncthreads();
#endif

    // prefetch next K-slab of the streamed A operand
    if (k0 + 32 < K) {
      int gr = tileM + (tid >> 1);
      if (gr < M) __builtin_prefetch(A + (long)gr * K + k0 + 32, 0, 1);
    }

    FragBF afr[2];
#pragma unroll
    for (int mt = 0; mt < 2; mt++) {
      int r  = mBase + mt * 16 + (lane & 15);
      int kA = (lane < 16) ? 0 : 8;   // ISA A-frag: lanes0-15 K0-7/16-23, lanes16-31 K8-15/24-31
      afr[mt].q[0] = *(const U4*)(As + r * 32 + kA);
      afr[mt].q[1] = *(const U4*)(As + r * 32 + kA + 16);
    }
#pragma unroll
    for (int nt = 0; nt < 4; nt++) {
      FragBF bfr;
      int n  = nBase + nt * 16 + (lane & 15);
      int kB = (lane < 16) ? 0 : 8;   // B-frag: lanes0-15 K0-7/16-23, lanes16-31 K8-15/24-31
      bfr.q[0] = *(const U4*)(Bs + n * 32 + kB);
      bfr.q[1] = *(const U4*)(Bs + n * 32 + kB + 16);
      acc[0][nt].v = __builtin_amdgcn_wmma_f32_16x16x32_bf16(
          false, afr[0].v, false, bfr.v, (short)0, acc[0][nt].v, false, false);
      acc[1][nt].v = __builtin_amdgcn_wmma_f32_16x16x32_bf16(
          false, afr[1].v, false, bfr.v, (short)0, acc[1][nt].v, false, false);
    }
    __syncthreads();
  }

  // epilogue: bias -> gelu -> residual; store f32 and/or bf16
#pragma unroll
  for (int mt = 0; mt < 2; mt++) {
#pragma unroll
    for (int nt = 0; nt < 4; nt++) {
      int col = tileN + nBase + nt * 16 + (lane & 15);
      float bval = bias ? bias[col] : 0.f;
#pragma unroll
      for (int j = 0; j < 8; j++) {
        int row = tileM + mBase + mt * 16 + j + ((lane < 16) ? 0 : 8);
        if (row < M) {
          long idx = (long)row * N + col;
          float x = acc[mt][nt].f[j] + bval;
          if (doGelu) x = 0.5f * x * (1.f + erff(x * 0.70710678118654752f));
          if (resid) x += resid[idx];
          if (outF) outF[idx] = x;
          if (outB) outB[idx] = f2bf(x);
        }
      }
    }
  }
}

// ---------------------------------------------------------------------------
// Flash attention: 1 wave per 16-query tile; 4 independent waves per block.
// Q(576xDK) K,V(640xDK) per (b,h); DK=32 = one bf16 WMMA K-step.
// Out(bf16) gets + pv broadcast for rows [64, 575)  (the q2/k2/v2 branch:
// softmax over a single key is identically 1, so that MHA collapses to a
// broadcast of pref_node @ Wv2 -- exact, not an approximation).
// ---------------------------------------------------------------------------
__global__ __launch_bounds__(128) void k_attn(
    const u16* __restrict__ Q, const u16* __restrict__ Km,
    const u16* __restrict__ Vm, const float* __restrict__ pv,
    u16* __restrict__ Out)
{
  __shared__ __attribute__((aligned(16))) u16 Ps[4 * 16 * 32];  // P tile per wave
  __shared__ __attribute__((aligned(16))) u16 Vs[4 * 32 * 40];  // V tile per wave (pad 40)

  const int lane = threadIdx.x & 31;
  const int wid  = threadIdx.x >> 5;
  const int gidx = blockIdx.x * 4 + wid;           // q-tile index: 32*16*36
  const int qt = gidx % 36;
  const int bh = gidx / 36;
  const int h  = bh & 15;
  const int b  = bh >> 4;
  const long qrow0 = (long)b * NSEQ + qt * 16;
  const long krow0 = (long)b * NKV;
  const int  hcol  = h * 32;

  u16* myPs = Ps + wid * (16 * 32);
  u16* myVs = Vs + wid * (32 * 40);

  // Q A-fragment (loaded once)
  FragBF qf;
  {
    int m  = lane & 15;
    int kA = (lane < 16) ? 0 : 8;
    const u16* qp = Q + (qrow0 + m) * DMODEL + hcol;
    qf.q[0] = *(const U4*)(qp + kA);
    qf.q[1] = *(const U4*)(qp + kA + 16);
  }

  FragF o0, o1, zf;
  float mrow[8], srow[8];
#pragma unroll
  for (int j = 0; j < 8; j++) {
    o0.f[j] = 0.f; o1.f[j] = 0.f; zf.f[j] = 0.f;
    mrow[j] = -1e30f; srow[j] = 0.f;
  }
  const float scale = 0.17677669529663687f;        // 1/sqrt(32)

  for (int kb = 0; kb < NKV; kb += 32) {
    // stage V tile (32 keys x 32 dims) in this wave's LDS slice
    {
      const u16* vp = Vm + (krow0 + kb + lane) * DMODEL + hcol;
      u16* dst = myVs + lane * 40;
#ifdef USE_ASYNC_LDS
      async_cp16(vp + 0,  dst + 0);
      async_cp16(vp + 8,  dst + 8);
      async_cp16(vp + 16, dst + 16);
      async_cp16(vp + 24, dst + 24);
      async_wait0();                  // per-wave LDS slice: no block barrier
#else
      U4 a0 = *(const U4*)(vp + 0);
      U4 a1 = *(const U4*)(vp + 8);
      U4 a2 = *(const U4*)(vp + 16);
      U4 a3 = *(const U4*)(vp + 24);
      *(U4*)(dst + 0)  = a0; *(U4*)(dst + 8)  = a1;
      *(U4*)(dst + 16) = a2; *(U4*)(dst + 24) = a3;
#endif
    }

    // S = Q @ K^T for 32 keys (two 16x16 tiles); K^T B-frag reads row-major K
    FragF s0, s1;
    {
      FragBF kf;
      int kl = lane & 15;
      int kS = (lane < 16) ? 0 : 8;
      const u16* kp0 = Km + (krow0 + kb + kl) * DMODEL + hcol + kS;
      kf.q[0] = *(const U4*)(kp0);
      kf.q[1] = *(const U4*)(kp0 + 16);
      s0.v = __builtin_amdgcn_wmma_f32_16x16x32_bf16(
          false, qf.v, false, kf.v, (short)0, zf.v, false, false);
      const u16* kp1 = Km + (krow0 + kb + 16 + kl) * DMODEL + hcol + kS;
      kf.q[0] = *(const U4*)(kp1);
      kf.q[1] = *(const U4*)(kp1 + 16);
      s1.v = __builtin_amdgcn_wmma_f32_16x16x32_bf16(
          false, qf.v, false, kf.v, (short)0, zf.v, false, false);
    }

    // online softmax (row reductions across 16 lanes of each half-wave)
#pragma unroll
    for (int j = 0; j < 8; j++) { s0.f[j] *= scale; s1.f[j] *= scale; }
#pragma unroll
    for (int j = 0; j < 8; j++) {
      float v = fmaxf(s0.f[j], s1.f[j]);
#pragma unroll
      for (int off = 8; off; off >>= 1) v = fmaxf(v, __shfl_xor(v, off, 16));
      float nm = fmaxf(mrow[j], v);
      float al = __expf(mrow[j] - nm);
      o0.f[j] *= al; o1.f[j] *= al;
      float e0 = __expf(s0.f[j] - nm);
      float e1 = __expf(s1.f[j] - nm);
      float rs = e0 + e1;
#pragma unroll
      for (int off = 8; off; off >>= 1) rs += __shfl_xor(rs, off, 16);
      srow[j] = srow[j] * al + rs;
      mrow[j] = nm;
      s0.f[j] = e0; s1.f[j] = e1;
    }

    // P (C-frag layout) -> LDS -> A-frag layout
    {
      int c = lane & 15;
#pragma unroll
      for (int j = 0; j < 8; j++) {
        int row = j + ((lane < 16) ? 0 : 8);
        myPs[row * 32 + c]      = f2bf(s0.f[j]);
        myPs[row * 32 + 16 + c] = f2bf(s1.f[j]);
      }
    }
    FragBF pf;
    {
      int m  = lane & 15;
      int kA = (lane < 16) ? 0 : 8;
      pf.q[0] = *(const U4*)(myPs + m * 32 + kA);
      pf.q[1] = *(const U4*)(myPs + m * 32 + kA + 16);
    }

    // V B-fragments (keys = K dim, dims = N) read transposed from LDS
    FragBF vf0, vf1;
    {
      int n  = lane & 15;
      int kS = (lane < 16) ? 0 : 8;
#pragma unroll
      for (int i = 0; i < 16; i++) {
        int kk = (i < 8) ? (kS + i) : (kS + 8 + i);
        vf0.h[i] = myVs[kk * 40 + n];
        vf1.h[i] = myVs[kk * 40 + 16 + n];
      }
    }
    o0.v = __builtin_amdgcn_wmma_f32_16x16x32_bf16(
        false, pf.v, false, vf0.v, (short)0, o0.v, false, false);
    o1.v = __builtin_amdgcn_wmma_f32_16x16x32_bf16(
        false, pf.v, false, vf1.v, (short)0, o1.v, false, false);
  }

  // finalize: O /= l, add pv broadcast for rows [64, 575), store bf16
  {
    int dim0 = lane & 15;
    float pv0 = pv[(long)b * DMODEL + hcol + dim0];
    float pv1 = pv[(long)b * DMODEL + hcol + 16 + dim0];
#pragma unroll
    for (int j = 0; j < 8; j++) {
      float inv = 1.f / srow[j];
      int q = qt * 16 + j + ((lane < 16) ? 0 : 8);
      float add = (q >= 64 && q < 575) ? 1.f : 0.f;
      long orow = ((long)b * NSEQ + q) * DMODEL + hcol;
      Out[orow + dim0]      = f2bf(o0.f[j] * inv + add * pv0);
      Out[orow + 16 + dim0] = f2bf(o1.f[j] * inv + add * pv1);
    }
  }
}

// ---------------------------------------------------------------------------
// Instance norm over the sequence axis (axis=1): per (b,d) column of 576.
// Optionally emits f32+bf16 full tensors, or scatters into split outputs.
// ---------------------------------------------------------------------------
__global__ __launch_bounds__(256) void k_inorm(
    const float* __restrict__ x, const float* __restrict__ gamma,
    const float* __restrict__ beta,
    float* __restrict__ outF, u16* __restrict__ outB,
    float* __restrict__ splitMain, float* __restrict__ splitPref)
{
  int i = blockIdx.x * 256 + threadIdx.x;
  if (i >= NBATCH * DMODEL) return;
  int d = i & (DMODEL - 1);
  int b = i / DMODEL;
  const float* xp = x + (long)b * NSEQ * DMODEL + d;
  float s = 0.f, s2 = 0.f;
  for (int n = 0; n < NSEQ; n++) {
    float v = xp[(long)n * DMODEL];
    s += v; s2 += v * v;
  }
  const float rN = 1.f / (float)NSEQ;
  float mean = s * rN;
  float var  = s2 * rN - mean * mean;
  float inv  = rsqrtf(var + 1e-5f);
  float g = gamma[d], be = beta[d];
  for (int n = 0; n < NSEQ; n++) {
    float y = (xp[(long)n * DMODEL] - mean) * inv * g + be;
    long idx = ((long)b * NSEQ + n) * DMODEL + d;
    if (outF) outF[idx] = y;
    if (outB) outB[idx] = f2bf(y);
    if (splitMain) {
      if (n >= 64) splitMain[((long)b * 512 + (n - 64)) * DMODEL + d] = y;
      else         splitPref[((long)b * 64  + n)        * DMODEL + d] = y;
    }
  }
}

// ---------------------------------------------------------------------------
// Host side
// ---------------------------------------------------------------------------
namespace {

struct Ws {
  float* in1_f;  u16* in1_b;  u16* kv_b;   u16* pref_b;
  u16* q1_b;     u16* k1_b;   u16* v1_b;   float* pv_f;
  u16* attn_b;   float* tmp_f; float* out1_f; u16* out1_b;
  u16* hidden_b;
  u16 *wq1t, *wk1t, *wv1t, *wv2t, *wct, *ff1t, *ff2t;
};

struct Branch {
  const float *inp, *fcp, *extra;
  const float *Wq1, *Wk1, *Wv1, *Wv2, *Wc, *Wcb, *F1, *F1b, *F2, *F2b;
  const float *n1w, *n1b, *n2w, *n2b;
  float *outMain, *outPref;
};

inline char* bump(size_t& o, char* base, size_t bytes) {
  o = (o + 255) & ~(size_t)255;
  char* p = base + o;
  o += bytes;
  return p;
}

void run_branch(const Branch& br, const Ws& w, hipStream_t stream) {
  const long M1 = (long)NBATCH * NSEQ;   // 18432
  const long Mk = (long)NBATCH * NKV;    // 20480

  // 1) inputs
  k_build_inputs<<<(NBATCH * NKV * DMODEL) / 256, 256, 0, stream>>>(
      br.fcp, br.inp, br.extra, w.in1_f, w.in1_b, w.kv_b);
  k_build_pref<<<(NBATCH * DMODEL) / 256, 256, 0, stream>>>(br.inp, w.pref_b);

  // 2) weight transpose+cast
  k_wt<<<(DMODEL * DMODEL) / 256, 256, 0, stream>>>(br.Wq1, w.wq1t, DMODEL, DMODEL);
  k_wt<<<(DMODEL * DMODEL) / 256, 256, 0, stream>>>(br.Wk1, w.wk1t, DMODEL, DMODEL);
  k_wt<<<(DMODEL * DMODEL) / 256, 256, 0, stream>>>(br.Wv1, w.wv1t, DMODEL, DMODEL);
  k_wt<<<(DMODEL * DMODEL) / 256, 256, 0, stream>>>(br.Wv2, w.wv2t, DMODEL, DMODEL);
  k_wt<<<(DMODEL * DMODEL) / 256, 256, 0, stream>>>(br.Wc,  w.wct,  DMODEL, DMODEL);
  k_wt<<<(DMODEL * NFF)    / 256, 256, 0, stream>>>(br.F1,  w.ff1t, DMODEL, NFF);
  k_wt<<<(NFF * DMODEL)    / 256, 256, 0, stream>>>(br.F2,  w.ff2t, NFF, DMODEL);

  // 3) projections (bf16 out) + pv = pref @ Wv2 (f32 out)
  k_gemm<<<dim3(DMODEL / 128, (unsigned)(M1 / 64)), 128, 0, stream>>>(
      w.in1_b, w.wq1t, nullptr, nullptr, nullptr, w.q1_b,
      (int)M1, DMODEL, DMODEL, 0);
  k_gemm<<<dim3(DMODEL / 128, (unsigned)(Mk / 64)), 128, 0, stream>>>(
      w.kv_b, w.wk1t, nullptr, nullptr, nullptr, w.k1_b,
      (int)Mk, DMODEL, DMODEL, 0);
  k_gemm<<<dim3(DMODEL / 128, (unsigned)(Mk / 64)), 128, 0, stream>>>(
      w.kv_b, w.wv1t, nullptr, nullptr, nullptr, w.v1_b,
      (int)Mk, DMODEL, DMODEL, 0);
  k_gemm<<<dim3(DMODEL / 128, 1), 128, 0, stream>>>(
      w.pref_b, w.wv2t, nullptr, nullptr, w.pv_f, nullptr,
      NBATCH, DMODEL, DMODEL, 0);

  // 4) flash attention (+pv fused)
  k_attn<<<(NBATCH * 16 * (NSEQ / 16)) / 4, 128, 0, stream>>>(
      w.q1_b, w.k1_b, w.v1_b, w.pv_f, w.attn_b);

  // 5) output projection + residual(input1) -> tmp ; inorm1 -> out1 (f32+bf16)
  k_gemm<<<dim3(DMODEL / 128, (unsigned)(M1 / 64)), 128, 0, stream>>>(
      w.attn_b, w.wct, br.Wcb, w.in1_f, w.tmp_f, nullptr,
      (int)M1, DMODEL, DMODEL, 0);
  k_inorm<<<(NBATCH * DMODEL) / 256, 256, 0, stream>>>(
      w.tmp_f, br.n1w, br.n1b, w.out1_f, w.out1_b, nullptr, nullptr);

  // 6) FFN: gelu(out1@ff1+b1) -> hidden(bf16) ; hidden@ff2+b2+out1 -> tmp
  k_gemm<<<dim3(NFF / 128, (unsigned)(M1 / 64)), 128, 0, stream>>>(
      w.out1_b, w.ff1t, br.F1b, nullptr, nullptr, w.hidden_b,
      (int)M1, NFF, DMODEL, 1);
  k_gemm<<<dim3(DMODEL / 128, (unsigned)(M1 / 64)), 128, 0, stream>>>(
      w.hidden_b, w.ff2t, br.F2b, w.out1_f, w.tmp_f, nullptr,
      (int)M1, DMODEL, NFF, 0);

  // 7) inorm2 with split store directly into d_out regions
  k_inorm<<<(NBATCH * DMODEL) / 256, 256, 0, stream>>>(
      w.tmp_f, br.n2w, br.n2b, nullptr, nullptr, br.outMain, br.outPref);
}

} // namespace

extern "C" void kernel_launch(void* const* d_in, const int* in_sizes, int n_in,
                              void* d_out, int out_size, void* d_ws, size_t ws_size,
                              hipStream_t stream) {
  (void)in_sizes; (void)n_in; (void)out_size; (void)ws_size;
  const float* input     = (const float*)d_in[0];
  const float* input_img = (const float*)d_in[1];
  const float* fcp       = (const float*)d_in[2];
  const float* fcp_img   = (const float*)d_in[3];
  const float* const* P  = (const float* const*)(d_in + 4); // 24 weight leaves then 8 norm leaves

  float* out = (float*)d_out;
  const long pMain0 = 0;
  const long pMain1 = (long)NBATCH * 512 * DMODEL;       // 8388608
  const long pPref0 = 2 * pMain1;                         // 16777216
  const long pPref1 = pPref0 + (long)NBATCH * 64 * DMODEL;

  // workspace carve-out (reused by both branches)
  char* base = (char*)d_ws;
  size_t o = 0;
  Ws w;
  w.in1_f    = (float*)bump(o, base, (size_t)NBATCH * NSEQ * DMODEL * 4);
  w.in1_b    = (u16*)  bump(o, base, (size_t)NBATCH * NSEQ * DMODEL * 2);
  w.kv_b     = (u16*)  bump(o, base, (size_t)NBATCH * NKV  * DMODEL * 2);
  w.pref_b   = (u16*)  bump(o, base, (size_t)NBATCH * DMODEL * 2);
  w.q1_b     = (u16*)  bump(o, base, (size_t)NBATCH * NSEQ * DMODEL * 2);
  w.k1_b     = (u16*)  bump(o, base, (size_t)NBATCH * NKV  * DMODEL * 2);
  w.v1_b     = (u16*)  bump(o, base, (size_t)NBATCH * NKV  * DMODEL * 2);
  w.pv_f     = (float*)bump(o, base, (size_t)NBATCH * DMODEL * 4);
  w.attn_b   = (u16*)  bump(o, base, (size_t)NBATCH * NSEQ * DMODEL * 2);
  w.tmp_f    = (float*)bump(o, base, (size_t)NBATCH * NSEQ * DMODEL * 4);
  w.out1_f   = (float*)bump(o, base, (size_t)NBATCH * NSEQ * DMODEL * 4);
  w.out1_b   = (u16*)  bump(o, base, (size_t)NBATCH * NSEQ * DMODEL * 2);
  w.hidden_b = (u16*)  bump(o, base, (size_t)NBATCH * NSEQ * NFF * 2);
  w.wq1t = (u16*)bump(o, base, (size_t)DMODEL * DMODEL * 2);
  w.wk1t = (u16*)bump(o, base, (size_t)DMODEL * DMODEL * 2);
  w.wv1t = (u16*)bump(o, base, (size_t)DMODEL * DMODEL * 2);
  w.wv2t = (u16*)bump(o, base, (size_t)DMODEL * DMODEL * 2);
  w.wct  = (u16*)bump(o, base, (size_t)DMODEL * DMODEL * 2);
  w.ff1t = (u16*)bump(o, base, (size_t)DMODEL * NFF * 2);
  w.ff2t = (u16*)bump(o, base, (size_t)NFF * DMODEL * 2);

  // params: per branch block of 12 leaves, then 8 norm leaves
  // [Wq1 Wk1 Wv1 Wq2 Wk2 Wv2 Wc_w Wc_b ff1_w ff1_b ff2_w ff2_b] x2, then
  // [n1_w n1_b n2_w n2_b] x2
  Branch b0, b1;
  b0.inp = input;     b0.fcp = fcp;     b0.extra = fcp_img;
  b0.Wq1 = P[0];  b0.Wk1 = P[1];  b0.Wv1 = P[2];  b0.Wv2 = P[5];
  b0.Wc  = P[6];  b0.Wcb = P[7];  b0.F1  = P[8];  b0.F1b = P[9];
  b0.F2  = P[10]; b0.F2b = P[11];
  b0.n1w = P[24]; b0.n1b = P[25]; b0.n2w = P[26]; b0.n2b = P[27];
  b0.outMain = out + pMain0; b0.outPref = out + pPref0;

  b1.inp = input_img; b1.fcp = fcp_img; b1.extra = fcp;
  b1.Wq1 = P[12]; b1.Wk1 = P[13]; b1.Wv1 = P[14]; b1.Wv2 = P[17];
  b1.Wc  = P[18]; b1.Wcb = P[19]; b1.F1  = P[20]; b1.F1b = P[21];
  b1.F2  = P[22]; b1.F2b = P[23];
  b1.n1w = P[28]; b1.n1b = P[29]; b1.n2w = P[30]; b1.n2b = P[31];
  b1.outMain = out + pMain1; b1.outPref = out + pPref1;

  run_branch(b0, w, stream);
  run_branch(b1, w, stream);
}
